// AttentionBlock_11819749998629
// MI455X (gfx1250) — matrix operand order repaired and checked by reading, not compiled, not run on hardware
//
#include <hip/hip_runtime.h>
#include <hip/hip_bf16.h>

#define BB 4
#define HH 64
#define WWD 64
#define CC 512
#define NN (HH*WWD)          // 4096 tokens per batch
#define MM (BB*NN)           // 16384 total rows
#define LN_EPS 1e-3f
#define QSCALE 0.044194173824159216f   // 1/sqrt(512)

typedef _Float16 v16h __attribute__((ext_vector_type(16)));
typedef _Float16 v8h  __attribute__((ext_vector_type(8)));
typedef float    v8f  __attribute__((ext_vector_type(8)));

// Fragment container. A-frag (16x32 f16): lane L holds row m=L%16,
// half h -> K = 16*g + h, g=L/16.
// B-frag (32x16 f16): lane L holds col n=L%16, half h -> K = 16*g + h.
union Frag { v16h v; v8h h[2]; };

// ---------------------------------------------------------------------------
// Async global->LDS copy of 16 bytes per lane (CDNA5 GLOBAL_LOAD_ASYNC_TO_LDS,
// tracked by ASYNCcnt). Falls back to a synchronous copy if the builtin is
// not available in this toolchain.
// ---------------------------------------------------------------------------
typedef int i4 __attribute__((vector_size(16)));
typedef __attribute__((address_space(1))) i4* gp_i4;   // global int4*
typedef __attribute__((address_space(3))) i4* lp_i4;   // LDS    int4*

__device__ __forceinline__ void async_copy16(const _Float16* gsrc, _Float16* ldst) {
#if __has_builtin(__builtin_amdgcn_global_load_async_to_lds_b128)
  __builtin_amdgcn_global_load_async_to_lds_b128(
      (gp_i4)(void*)gsrc, (lp_i4)(void*)ldst, 0, 0);
#else
  *(v8h*)ldst = *(const v8h*)gsrc;
#endif
}

__device__ __forceinline__ void wait_async0() {
#if __has_builtin(__builtin_amdgcn_global_load_async_to_lds_b128)
#if __has_builtin(__builtin_amdgcn_s_wait_asynccnt)
  __builtin_amdgcn_s_wait_asynccnt(0);
#else
  asm volatile("s_wait_asynccnt 0" ::: "memory");
#endif
#endif
}

// ---------------------------------------------------------------------------
// Kernel 0: convert 4 weight matrices fp32 [C,C] -> f16 transposed [Cout,Cin]
// so WMMA B-operand loads (lane = out column, 16 contiguous K halfs) coalesce.
// ---------------------------------------------------------------------------
__global__ void wconv_kernel(const float* __restrict__ w0, const float* __restrict__ w1,
                             const float* __restrict__ w2, const float* __restrict__ w3,
                             _Float16* __restrict__ o0, _Float16* __restrict__ o1,
                             _Float16* __restrict__ o2, _Float16* __restrict__ o3) {
  int z = blockIdx.y;
  const float* src = (z == 0) ? w0 : (z == 1) ? w1 : (z == 2) ? w2 : w3;
  _Float16* dst    = (z == 0) ? o0 : (z == 1) ? o1 : (z == 2) ? o2 : o3;
  int i = blockIdx.x * blockDim.x + threadIdx.x;     // over C*C
  int j = i / CC;                                    // out column
  int c = i % CC;                                    // in channel
  dst[(size_t)j * CC + c] = (_Float16)src[(size_t)c * CC + j];
}

// ---------------------------------------------------------------------------
// Kernel 1: LayerNorm over C, write f16 hn. One block (128 thr) per row.
// ---------------------------------------------------------------------------
__global__ __launch_bounds__(128) void ln_kernel(const float* __restrict__ x,
                                                 const float* __restrict__ gam,
                                                 const float* __restrict__ bet,
                                                 _Float16* __restrict__ hn) {
  int row = blockIdx.x;
  int tid = threadIdx.x;                 // 128 threads, 4 channels each
  const float4 v = ((const float4*)(x + (size_t)row * CC))[tid];
  float s  = v.x + v.y + v.z + v.w;
  float s2 = v.x * v.x + v.y * v.y + v.z * v.z + v.w * v.w;
  __shared__ float r1[128], r2[128];
  r1[tid] = s; r2[tid] = s2;
  __syncthreads();
  for (int st = 64; st > 0; st >>= 1) {
    if (tid < st) { r1[tid] += r1[tid + st]; r2[tid] += r2[tid + st]; }
    __syncthreads();
  }
  float mean = r1[0] * (1.0f / CC);
  float var  = r2[0] * (1.0f / CC) - mean * mean;
  float rstd = rsqrtf(var + LN_EPS);
  const float4 g4 = ((const float4*)gam)[tid];
  const float4 b4 = ((const float4*)bet)[tid];
  _Float16* o = hn + (size_t)row * CC + tid * 4;
  o[0] = (_Float16)((v.x - mean) * rstd * g4.x + b4.x);
  o[1] = (_Float16)((v.y - mean) * rstd * g4.y + b4.y);
  o[2] = (_Float16)((v.z - mean) * rstd * g4.z + b4.z);
  o[3] = (_Float16)((v.w - mean) * rstd * g4.w + b4.w);
}

// ---------------------------------------------------------------------------
// Kernel 2: QKV projections. grid.z selects Wq/Wk/Wv. Each wave: 16x64 tile.
// Q scaled by 1/sqrt(C); V stored transposed [B, C, N] for the PV WMMA.
// ---------------------------------------------------------------------------
__global__ __launch_bounds__(128) void qkv_kernel(
    const _Float16* __restrict__ hn,
    const _Float16* __restrict__ wqT, const _Float16* __restrict__ wkT,
    const _Float16* __restrict__ wvT,
    const float* __restrict__ bq, const float* __restrict__ bk,
    const float* __restrict__ bvv,
    _Float16* __restrict__ qo, _Float16* __restrict__ ko,
    _Float16* __restrict__ vto) {
  int mode = blockIdx.z;
  const _Float16* wt  = (mode == 0) ? wqT : (mode == 1) ? wkT : wvT;
  const float* bias   = (mode == 0) ? bq  : (mode == 1) ? bk  : bvv;
  int tid = threadIdx.x, w = tid >> 5, lane = tid & 31, n = lane & 15, g = lane >> 4;
  int row0 = blockIdx.x * 64 + w * 16;
  int col0 = blockIdx.y * 64;
  v8f acc[4] = {};
  const _Float16* abase = hn + (size_t)(row0 + n) * CC;
  #pragma unroll 4
  for (int kc = 0; kc < CC; kc += 32) {
    Frag a;
    a.h[0] = *(const v8h*)(abase + kc + 16 * g);
    a.h[1] = *(const v8h*)(abase + kc + 16 * g + 8);
    #pragma unroll
    for (int j = 0; j < 4; ++j) {
      int col = col0 + 16 * j + n;
      v16h bfr = *(const v16h*)(wt + (size_t)col * CC + kc + 16 * g);
      acc[j] = __builtin_amdgcn_wmma_f32_16x16x32_f16(false, a.v, false, bfr,
                                                      (short)0, acc[j], false, false);
    }
  }
  #pragma unroll
  for (int j = 0; j < 4; ++j) {
    int col = col0 + 16 * j + n;
    float bsv = bias[col];
    #pragma unroll
    for (int r = 0; r < 8; ++r) {
      int row = row0 + 8 * g + r;
      float val = acc[j][r] + bsv;
      if (mode == 0) {
        qo[(size_t)row * CC + col] = (_Float16)(val * QSCALE);
      } else if (mode == 1) {
        ko[(size_t)row * CC + col] = (_Float16)val;
      } else {
        int bidx = row >> 12;            // row / NN
        int nn   = row & (NN - 1);
        vto[((size_t)bidx * CC + col) * NN + nn] = (_Float16)val;
      }
    }
  }
}

// ---------------------------------------------------------------------------
// Kernel 3: fused flash attention, L2-traffic-optimized.
// Block = 8 waves (256 thr), 64 queries. wave = (qtile 0..3) x (chan-half 0..1).
// Per 64-key chunk: K (64KB, contiguous) and V^T slices (64KB) are staged
// into LDS with async global->LDS copies; each wave then computes its
// 16q x 32k score subtile (32 WMMAs) and its 16q x 256ch PV slice (32 WMMAs)
// entirely from LDS. Block-wide online softmax via shuffles + LDS stats.
// ---------------------------------------------------------------------------
#define KP (512 + 8)   // Klds row pitch (halfs): 1040B -> bank stride 4
#define VP (64 + 8)    // Vlds/Plds row pitch (halfs): 144B -> 36-dword row stride
__global__ __launch_bounds__(256) void attn_kernel(
    const _Float16* __restrict__ qg, const _Float16* __restrict__ kg,
    const _Float16* __restrict__ vtg, _Float16* __restrict__ og) {
  int b  = blockIdx.y;
  int q0 = blockIdx.x * 64;
  int tid = threadIdx.x;
  int wv  = tid >> 5;          // 0..7
  int qt  = wv >> 1;           // query subtile 0..3
  int ch  = wv & 1;            // channel half 0..1
  int lane = tid & 31, n = lane & 15, g = lane >> 4;

  __shared__ __align__(16) _Float16 Klds[64][KP];   // K rows  [key][c]
  __shared__ __align__(16) _Float16 Vlds[512][VP];  // V^T     [c][key]
  __shared__ __align__(16) _Float16 Plds[64][VP];   // softmax [q][key]
  __shared__ float smax[4][2][16];
  __shared__ float ssum[4][2][16];

  // Preload this wave's Q tile (rows q0+qt*16 .. +15) as A-fragments.
  Frag qa[16];
  const _Float16* qbase = qg + (size_t)(b * NN + q0 + qt * 16 + n) * CC;
  #pragma unroll
  for (int kc2 = 0; kc2 < 16; ++kc2) {
    qa[kc2].h[0] = *(const v8h*)(qbase + kc2 * 32 + 16 * g);
    qa[kc2].h[1] = *(const v8h*)(qbase + kc2 * 32 + 16 * g + 8);
  }

  float mr[8], lr[8];
  v8f acc[16] = {};
  #pragma unroll
  for (int r = 0; r < 8; ++r) { mr[r] = -3.0e38f; lr[r] = 0.0f; }

  const _Float16* kgbase = kg + (size_t)(b * NN) * CC;
  const _Float16* vtbase = vtg + (size_t)b * CC * NN;

  for (int k0 = 0; k0 < NN; k0 += 64) {
    __syncthreads();   // previous chunk's LDS consumers are done
    // ---- stage K chunk (contiguous 64KB) into LDS ----
    {
      const _Float16* src = kgbase + (size_t)k0 * CC;
      #pragma unroll
      for (int it = 0; it < 16; ++it) {
        int idx  = it * 256 + tid;        // 4096 x 16B pieces
        int row  = idx >> 6;              // key row 0..63
        int coff = (idx & 63) * 8;        // half offset in row
        async_copy16(src + row * CC + coff, &Klds[row][coff]);
      }
      // ---- stage V^T chunk (512 rows x 128B) into LDS ----
      #pragma unroll
      for (int it = 0; it < 16; ++it) {
        int idx = it * 256 + tid;         // 4096 x 16B pieces
        int c   = idx >> 3;               // channel 0..511
        int ko  = (idx & 7) * 8;          // key offset 0..56
        async_copy16(vtbase + (size_t)c * NN + k0 + ko, &Vlds[c][ko]);
      }
      wait_async0();
    }
    __syncthreads();

    // ---- scores: this wave's 16q x 32k (two 16x16 tiles), K from LDS ----
    v8f s0 = {}, s1 = {};
    #pragma unroll
    for (int kc2 = 0; kc2 < 16; ++kc2) {
      Frag b0, b1;
      const _Float16* kr0 = &Klds[ch * 32 + n][kc2 * 32 + 16 * g];
      const _Float16* kr1 = &Klds[ch * 32 + 16 + n][kc2 * 32 + 16 * g];
      b0.h[0] = *(const v8h*)(kr0);  b0.h[1] = *(const v8h*)(kr0 + 8);
      b1.h[0] = *(const v8h*)(kr1);  b1.h[1] = *(const v8h*)(kr1 + 8);
      s0 = __builtin_amdgcn_wmma_f32_16x16x32_f16(false, qa[kc2].v, false, b0.v,
                                                  (short)0, s0, false, false);
      s1 = __builtin_amdgcn_wmma_f32_16x16x32_f16(false, qa[kc2].v, false, b1.v,
                                                  (short)0, s1, false, false);
    }
    // ---- per-row max over this wave's 32 keys ----
    float vmax[8];
    #pragma unroll
    for (int r = 0; r < 8; ++r) {
      float v = fmaxf(s0[r], s1[r]);
      v = fmaxf(v, __shfl_xor(v, 1, 32));
      v = fmaxf(v, __shfl_xor(v, 2, 32));
      v = fmaxf(v, __shfl_xor(v, 4, 32));
      v = fmaxf(v, __shfl_xor(v, 8, 32));
      vmax[r] = v;
    }
    if (n == 0) {
      #pragma unroll
      for (int r = 0; r < 8; ++r) smax[qt][ch][8 * g + r] = vmax[r];
    }
    __syncthreads();
    // ---- online softmax update (block max over 64 keys) ----
    float mnew[8], alpha[8];
    #pragma unroll
    for (int r = 0; r < 8; ++r) {
      int rowi = 8 * g + r;
      float cm = fmaxf(mr[r], fmaxf(smax[qt][0][rowi], smax[qt][1][rowi]));
      mnew[r]  = cm;
      alpha[r] = __expf(mr[r] - cm);
      mr[r]    = cm;
    }
    #pragma unroll
    for (int j = 0; j < 16; ++j) {
      #pragma unroll
      for (int r = 0; r < 8; ++r) acc[j][r] *= alpha[r];
    }
    float rs[8];
    #pragma unroll
    for (int r = 0; r < 8; ++r) {
      float p0 = __expf(s0[r] - mnew[r]);
      float p1 = __expf(s1[r] - mnew[r]);
      s0[r] = p0; s1[r] = p1;
      float v = p0 + p1;
      v += __shfl_xor(v, 1, 32);
      v += __shfl_xor(v, 2, 32);
      v += __shfl_xor(v, 4, 32);
      v += __shfl_xor(v, 8, 32);
      rs[r] = v;
    }
    #pragma unroll
    for (int r = 0; r < 8; ++r) {
      Plds[qt * 16 + 8 * g + r][ch * 32 + n]      = (_Float16)s0[r];
      Plds[qt * 16 + 8 * g + r][ch * 32 + 16 + n] = (_Float16)s1[r];
    }
    if (n == 0) {
      #pragma unroll
      for (int r = 0; r < 8; ++r) ssum[qt][ch][8 * g + r] = rs[r];
    }
    __syncthreads();
    #pragma unroll
    for (int r = 0; r < 8; ++r) {
      int rowi = 8 * g + r;
      lr[r] = lr[r] * alpha[r] + ssum[qt][0][rowi] + ssum[qt][1][rowi];
    }
    // ---- PV: this wave's 16q x 256ch slice, P and V from LDS ----
    #pragma unroll
    for (int kc2 = 0; kc2 < 2; ++kc2) {
      Frag pa;
      const _Float16* pr = &Plds[qt * 16 + n][0];
      pa.h[0] = *(const v8h*)(pr + kc2 * 32 + 16 * g);
      pa.h[1] = *(const v8h*)(pr + kc2 * 32 + 16 * g + 8);
      #pragma unroll
      for (int j = 0; j < 16; ++j) {
        int c = ch * 256 + 16 * j + n;
        Frag bf;
        const _Float16* vr = &Vlds[c][kc2 * 32 + 16 * g];
        bf.h[0] = *(const v8h*)(vr);
        bf.h[1] = *(const v8h*)(vr + 8);
        acc[j] = __builtin_amdgcn_wmma_f32_16x16x32_f16(false, pa.v, false, bf.v,
                                                        (short)0, acc[j], false, false);
      }
    }
  }
  // ---- normalize and store O (f16) ----
  #pragma unroll
  for (int j = 0; j < 16; ++j) {
    int c = ch * 256 + 16 * j + n;
    #pragma unroll
    for (int r = 0; r < 8; ++r) {
      int row = q0 + qt * 16 + 8 * g + r;
      og[(size_t)(b * NN + row) * CC + c] = (_Float16)(acc[j][r] / lr[r]);
    }
  }
}

// ---------------------------------------------------------------------------
// Kernel 4: output projection + bias + residual (fp32 out).
// ---------------------------------------------------------------------------
__global__ __launch_bounds__(128) void proj_kernel(
    const _Float16* __restrict__ og, const _Float16* __restrict__ wpT,
    const float* __restrict__ bp, const float* __restrict__ x,
    float* __restrict__ out) {
  int tid = threadIdx.x, w = tid >> 5, lane = tid & 31, n = lane & 15, g = lane >> 4;
  int row0 = blockIdx.x * 64 + w * 16;
  int col0 = blockIdx.y * 64;
  v8f acc[4] = {};
  const _Float16* abase = og + (size_t)(row0 + n) * CC;
  #pragma unroll 4
  for (int kc = 0; kc < CC; kc += 32) {
    Frag a;
    a.h[0] = *(const v8h*)(abase + kc + 16 * g);
    a.h[1] = *(const v8h*)(abase + kc + 16 * g + 8);
    #pragma unroll
    for (int j = 0; j < 4; ++j) {
      int col = col0 + 16 * j + n;
      v16h bfr = *(const v16h*)(wpT + (size_t)col * CC + kc + 16 * g);
      acc[j] = __builtin_amdgcn_wmma_f32_16x16x32_f16(false, a.v, false, bfr,
                                                      (short)0, acc[j], false, false);
    }
  }
  #pragma unroll
  for (int j = 0; j < 4; ++j) {
    int col = col0 + 16 * j + n;
    float bb = bp[col];
    #pragma unroll
    for (int r = 0; r < 8; ++r) {
      size_t idx = (size_t)(row0 + 8 * g + r) * CC + col;
      out[idx] = x[idx] + acc[j][r] + bb;
    }
  }
}

// ---------------------------------------------------------------------------
extern "C" void kernel_launch(void* const* d_in, const int* in_sizes, int n_in,
                              void* d_out, int out_size, void* d_ws, size_t ws_size,
                              hipStream_t stream) {
  (void)in_sizes; (void)n_in; (void)out_size; (void)ws_size;
  const float* x    = (const float*)d_in[0];
  const float* ln_g = (const float*)d_in[1];
  const float* ln_b = (const float*)d_in[2];
  const float* Wq   = (const float*)d_in[3];
  const float* bq   = (const float*)d_in[4];
  const float* Wk   = (const float*)d_in[5];
  const float* bk   = (const float*)d_in[6];
  const float* Wv   = (const float*)d_in[7];
  const float* bv   = (const float*)d_in[8];
  const float* Wp   = (const float*)d_in[9];
  const float* bp   = (const float*)d_in[10];
  float* out = (float*)d_out;

  char* ws = (char*)d_ws;
  const size_t MB = 1024u * 1024u;
  _Float16* hn  = (_Float16*)(ws + 0 * MB);    // 16 MB  [M, C]
  _Float16* qb  = (_Float16*)(ws + 16 * MB);   // 16 MB  [M, C] (pre-scaled)
  _Float16* kb  = (_Float16*)(ws + 32 * MB);   // 16 MB  [M, C]
  _Float16* vtb = (_Float16*)(ws + 48 * MB);   // 16 MB  [B, C, N] (V transposed)
  _Float16* ob  = (_Float16*)(ws + 64 * MB);   // 16 MB  [M, C]
  _Float16* wqT = (_Float16*)(ws + 80 * MB);   // 512 KB each, transposed f16
  _Float16* wkT = (_Float16*)(ws + 80 * MB + 512 * 1024);
  _Float16* wvT = (_Float16*)(ws + 81 * MB);
  _Float16* wpT = (_Float16*)(ws + 81 * MB + 512 * 1024);

  wconv_kernel<<<dim3(CC * CC / 256, 4), 256, 0, stream>>>(Wq, Wk, Wv, Wp,
                                                           wqT, wkT, wvT, wpT);
  ln_kernel<<<dim3(MM), 128, 0, stream>>>(x, ln_g, ln_b, hn);
  qkv_kernel<<<dim3(MM / 64, CC / 64, 3), 128, 0, stream>>>(hn, wqT, wkT, wvT,
                                                            bq, bk, bv, qb, kb, vtb);
  attn_kernel<<<dim3(NN / 64, BB), 256, 0, stream>>>(qb, kb, vtb, ob);
  proj_kernel<<<dim3(MM / 64, CC / 64), 128, 0, stream>>>(ob, wpT, bp, x, out);
}
